// SimpleAttention_53412213293575
// MI455X (gfx1250) — matrix operand order repaired and checked by reading, not compiled, not run on hardware
//
#include <hip/hip_runtime.h>
#include <hip/hip_bf16.h>
#include <stdint.h>

// Problem constants (match reference)
#define HIDDEN 1024
#define BATCH  4
#define SEQ    2048

// LDS row pitch in bf16 elements. 40 elems = 80B: row-indexed ds_load_b128
// start banks walk 0,20,40,60,16,36,... (stride 20 mod 64) -> 16 lanes hit
// disjoint 4-bank groups, no conflicts. 80B is 16B-aligned for b128 ops.
#define LDS_PITCH 40

typedef __attribute__((ext_vector_type(16))) __bf16 v16bf;
typedef __attribute__((ext_vector_type(8)))  float  v8f;
typedef unsigned long long ull;

struct Frag { uint32_t r[8]; };   // 8 VGPRs, 16 packed bf16

// ---- f32 -> bf16 pack (round to nearest even) ----
__device__ __forceinline__ uint32_t pack2_bf16(float lo, float hi) {
  uint32_t a = __float_as_uint(lo), b = __float_as_uint(hi);
  a += 0x7FFFu + ((a >> 16) & 1u);
  b += 0x7FFFu + ((b >> 16) & 1u);
  return (a >> 16) | (b & 0xFFFF0000u);
}
__device__ __forceinline__ uint16_t f32_to_bf16(float x) {
  uint32_t u = __float_as_uint(x);
  u += 0x7FFFu + ((u >> 16) & 1u);
  return (uint16_t)(u >> 16);
}

// ---- CDNA5 async memory->LDS copy (ASYNCcnt-tracked, no VGPR round-trip) ----
// VGLOBAL encoding: VDST = per-lane LDS byte address, VADDR = 64-bit global addr.
__device__ __forceinline__ void async_cp16(uint32_t lds_byte_off, const void* g) {
  asm volatile("global_load_async_to_lds_b128 %0, %1, off"
               :: "v"(lds_byte_off), "v"((unsigned long long)(uintptr_t)g)
               : "memory");
}
__device__ __forceinline__ void wait_async0() {
#if __has_builtin(__builtin_amdgcn_s_wait_asynccnt)
  __builtin_amdgcn_s_wait_asynccnt(0);
#else
  asm volatile("s_wait_asynccnt 0x0" ::: "memory");
#endif
}

// ---- bulk f32 -> bf16 conversion, 8 elements/thread, vector I/O ----
__global__ __launch_bounds__(256)
void cvt_f32_bf16(const float* __restrict__ src, uint16_t* __restrict__ dst, int n8) {
  const size_t i = (size_t)blockIdx.x * 256 + threadIdx.x;
  if (i >= (size_t)n8) return;
  const float4* s = (const float4*)src + i * 2;
  float4 a = s[0], b = s[1];
  uint4 o;
  o.x = pack2_bf16(a.x, a.y); o.y = pack2_bf16(a.z, a.w);
  o.z = pack2_bf16(b.x, b.y); o.w = pack2_bf16(b.z, b.w);
  ((uint4*)dst)[i] = o;
}

// ---- bf16 NT GEMM, async-LDS double-buffered operand staging ----
// Out[b] = (A[b] @ B[b]^T) * scale (+ bias)
// WG = 256 threads (8 waves). Macro-tile 64(M) x 256(N), k-step 32.
// Wave: 32x64 via 2 A-frags x 4 B-frags = 8 WMMA per k-step.
// Tiles move global->LDS via GLOBAL_LOAD_ASYNC_TO_LDS_B128 into buffer (i+1)
// while WMMAs consume buffer i; one barrier + s_wait_asynccnt per k-step.
template <bool OUTF32, bool TRANS_OUT, bool HAS_BIAS>
__global__ __launch_bounds__(256)
void gemm_bf16_nt(const uint16_t* __restrict__ A, const uint16_t* __restrict__ Bm,
                  const float* __restrict__ bias, void* __restrict__ Out,
                  int K, int ldc, float scale, ull sA, ull sB, ull sO) {
  // Single merged shared block so LDS byte offsets are known (block base = 0).
  // Per buffer: A tile 64 rows, then B tile 256 rows; row pitch LDS_PITCH.
  __shared__ uint16_t smem[2 * 320 * LDS_PITCH];   // 2 x 25600 B
  const uint32_t ABYTES   = 64u * LDS_PITCH * 2u;  //  5120 B (A region size)
  const uint32_t BUFBYTES = 320u * LDS_PITCH * 2u; // 25600 B (full buffer)

  const int tid  = threadIdx.x;
  const int lane = tid & 31;
  const int wave = tid >> 5;
  const int m0g = blockIdx.x * 64;
  const int n0g = blockIdx.y * 256;
  const int wm = (wave & 1) * 32;
  const int wn = (wave >> 1) * 64;
  const size_t bz = blockIdx.z;

  const uint16_t* Ap = A + bz * sA;
  const uint16_t* Bp = Bm + bz * sB;
  char*           Op = (char*)Out + bz * sO * (OUTF32 ? 4 : 2);

  // Staging assignment: A tile 64 rows x 32k (4 segs of 8 elems/row),
  // B tile 256 rows x 32k as 4 row-blocks of 64. 5 async b128 per thread.
  const int arow = tid >> 2, aseg = tid & 3;
  const uint16_t* agp = Ap + (size_t)(m0g + arow) * K + aseg * 8;
  const uint16_t* bgp = Bp + (size_t)(n0g + arow) * K + aseg * 8;
  const size_t bK64 = (size_t)64 * K;

  const uint32_t aoff  = (uint32_t)(arow * LDS_PITCH + aseg * 8) * 2u;
  const uint32_t boff  = ABYTES + aoff;
  const uint32_t brstep = 64u * LDS_PITCH * 2u;

  // Issue k-step 0 into buffer 0.
  async_cp16(aoff, agp);
  async_cp16(boff,              bgp);
  async_cp16(boff +     brstep, bgp + bK64);
  async_cp16(boff + 2 * brstep, bgp + 2 * bK64);
  async_cp16(boff + 3 * brstep, bgp + 3 * bK64);

  v8f acc[2][4];
  const v8f vzero = {0.f, 0.f, 0.f, 0.f, 0.f, 0.f, 0.f, 0.f};
#pragma unroll
  for (int mt = 0; mt < 2; ++mt)
#pragma unroll
    for (int t = 0; t < 4; ++t) acc[mt][t] = vzero;

  const int lmn = lane & 15;   // fragment M (for A) / N (for B) lane index
  const int lh  = lane >> 4;   // fragment K-half selector
  uint32_t buf = 0;            // current buffer byte offset (0 or BUFBYTES)

  for (int kb = 0; kb < K; kb += 32) {
    wait_async0();     // own async copies into current buffer complete
    __syncthreads();   // everyone's copies complete / prior reads done

    if (kb + 32 < K) { // stream next tile into the other buffer
      const int ko = kb + 32;
      const uint32_t nb = buf ^ BUFBYTES;
      async_cp16(nb + aoff, agp + ko);
      async_cp16(nb + boff,              bgp + ko);
      async_cp16(nb + boff +     brstep, bgp + bK64 + ko);
      async_cp16(nb + boff + 2 * brstep, bgp + 2 * bK64 + ko);
      async_cp16(nb + boff + 3 * brstep, bgp + 3 * bK64 + ko);
    }

    const uint16_t* smA = smem + (buf >> 1);           // element index = bytes/2
    const uint16_t* smB = smA + 64 * LDS_PITCH;

    // A fragments (ISA 7.12.2 16-bit A 16x32): lane m=lmn, half=lh;
    // VGPR0..3 = K 8h+{0..7}, VGPR4..7 = K 16+8h+{0..7}.
    Frag fa0, fa1;
    {
      const uint16_t* pa = smA + (size_t)(wm + lmn) * LDS_PITCH + 8 * lh;
      uint4 a = ((const uint4*)pa)[0];
      uint4 b = ((const uint4*)pa)[2];  // +16 elems
      fa0.r[0]=a.x; fa0.r[1]=a.y; fa0.r[2]=a.z; fa0.r[3]=a.w;
      fa0.r[4]=b.x; fa0.r[5]=b.y; fa0.r[6]=b.z; fa0.r[7]=b.w;
      pa += (size_t)16 * LDS_PITCH;
      a = ((const uint4*)pa)[0];
      b = ((const uint4*)pa)[2];
      fa1.r[0]=a.x; fa1.r[1]=a.y; fa1.r[2]=a.z; fa1.r[3]=a.w;
      fa1.r[4]=b.x; fa1.r[5]=b.y; fa1.r[6]=b.z; fa1.r[7]=b.w;
    }
    v16bf a0 = __builtin_bit_cast(v16bf, fa0);
    v16bf a1 = __builtin_bit_cast(v16bf, fa1);

#pragma unroll
    for (int t = 0; t < 4; ++t) {
      // B fragment (32x16 from row-major [N,K] tile): lane n=lmn, khalf=lh;
      // VGPR0..3 = K 8h+{0..7}, VGPR4..7 = K 16+8h+{0..7} (same as A).
      const uint16_t* pb = smB + (size_t)(wn + 16 * t + lmn) * LDS_PITCH + 8 * lh;
      uint4 x0 = ((const uint4*)pb)[0];
      uint4 x1 = ((const uint4*)pb)[2];
      Frag fb;
      fb.r[0]=x0.x; fb.r[1]=x0.y; fb.r[2]=x0.z; fb.r[3]=x0.w;
      fb.r[4]=x1.x; fb.r[5]=x1.y; fb.r[6]=x1.z; fb.r[7]=x1.w;
      v16bf bvv = __builtin_bit_cast(v16bf, fb);
      acc[0][t] = __builtin_amdgcn_wmma_f32_16x16x32_bf16(
          false, a0, false, bvv, (short)0, acc[0][t], false, false);
      acc[1][t] = __builtin_amdgcn_wmma_f32_16x16x32_bf16(
          false, a1, false, bvv, (short)0, acc[1][t], false, false);
    }
    buf ^= BUFBYTES;
  }

  // Epilogue: C/D layout (ISA 7.12.2): VGPR r -> (M = r + 8*(lane>>4), N = lane&15)
#pragma unroll
  for (int mt = 0; mt < 2; ++mt) {
#pragma unroll
    for (int t = 0; t < 4; ++t) {
      const int n = n0g + wn + 16 * t + lmn;
      const float bval = HAS_BIAS ? bias[n] : 0.0f;
#pragma unroll
      for (int r = 0; r < 8; ++r) {
        const int m = m0g + wm + 16 * mt + r + 8 * lh;
        const float v = acc[mt][t][r] * scale + bval;
        const size_t idx = TRANS_OUT ? (size_t)n * ldc + m : (size_t)m * ldc + n;
        if constexpr (OUTF32) ((float*)Op)[idx] = v;
        else                  ((uint16_t*)Op)[idx] = f32_to_bf16(v);
      }
    }
  }
}

// ---- Row softmax over SEQ f32 logits -> bf16 probabilities ----
__global__ __launch_bounds__(256)
void softmax_rows(const float* __restrict__ S, uint16_t* __restrict__ P) {
  const size_t row = blockIdx.x;
  const int tid = threadIdx.x;
  const float4* s4 = (const float4*)(S + row * (size_t)SEQ);
  float4 v0 = s4[tid * 2];
  float4 v1 = s4[tid * 2 + 1];
  __shared__ float red[256];

  float mx = fmaxf(fmaxf(fmaxf(v0.x, v0.y), fmaxf(v0.z, v0.w)),
                   fmaxf(fmaxf(v1.x, v1.y), fmaxf(v1.z, v1.w)));
  red[tid] = mx;
  __syncthreads();
#pragma unroll
  for (int off = 128; off > 0; off >>= 1) {
    if (tid < off) red[tid] = fmaxf(red[tid], red[tid + off]);
    __syncthreads();
  }
  mx = red[0];
  __syncthreads();

  float e[8];
  e[0] = __expf(v0.x - mx); e[1] = __expf(v0.y - mx);
  e[2] = __expf(v0.z - mx); e[3] = __expf(v0.w - mx);
  e[4] = __expf(v1.x - mx); e[5] = __expf(v1.y - mx);
  e[6] = __expf(v1.z - mx); e[7] = __expf(v1.w - mx);
  float sum = ((e[0] + e[1]) + (e[2] + e[3])) + ((e[4] + e[5]) + (e[6] + e[7]));
  red[tid] = sum;
  __syncthreads();
#pragma unroll
  for (int off = 128; off > 0; off >>= 1) {
    if (tid < off) red[tid] += red[tid + off];
    __syncthreads();
  }
  const float rinv = 1.0f / red[0];

  uint4 o;
  o.x = pack2_bf16(e[0] * rinv, e[1] * rinv);
  o.y = pack2_bf16(e[2] * rinv, e[3] * rinv);
  o.z = pack2_bf16(e[4] * rinv, e[5] * rinv);
  o.w = pack2_bf16(e[6] * rinv, e[7] * rinv);
  ((uint4*)(P + row * (size_t)SEQ))[tid] = o;
}

extern "C" void kernel_launch(void* const* d_in, const int* in_sizes, int n_in,
                              void* d_out, int out_size, void* d_ws, size_t ws_size,
                              hipStream_t stream) {
  (void)in_sizes; (void)n_in; (void)out_size; (void)ws_size;

  const float* x  = (const float*)d_in[0];
  const float* Wq = (const float*)d_in[1];
  const float* bq = (const float*)d_in[2];
  const float* Wk = (const float*)d_in[3];
  const float* bk = (const float*)d_in[4];
  const float* Wv = (const float*)d_in[5];
  const float* bv = (const float*)d_in[6];
  const float* Wo = (const float*)d_in[7];
  const float* bo = (const float*)d_in[8];
  float* out = (float*)d_out;

  // Workspace layout (184 MB total):
  //   xb bf16 [B][S][H]:16MB  Wqb/Wkb/Wvb/Wob bf16 [H][H]:2MB each
  //   Qb,Kb bf16 [B][S][H]:16MB each   Vt bf16 [B][H][S]:16MB
  //   Sb f32 [B][S][S]:64MB  Pb bf16 [B][S][S]:32MB  Cb bf16 [B][S][H]:16MB
  char* ws = (char*)d_ws;
  const size_t SZ_X  = (size_t)BATCH * SEQ * HIDDEN * 2;
  const size_t SZ_W  = (size_t)HIDDEN * HIDDEN * 2;
  const size_t SZ_SF = (size_t)BATCH * SEQ * SEQ * 4;
  const size_t SZ_SB = (size_t)BATCH * SEQ * SEQ * 2;
  uint16_t* xb  = (uint16_t*)(ws);
  uint16_t* Wqb = (uint16_t*)(ws + SZ_X);
  uint16_t* Wkb = (uint16_t*)(ws + SZ_X + 1 * SZ_W);
  uint16_t* Wvb = (uint16_t*)(ws + SZ_X + 2 * SZ_W);
  uint16_t* Wob = (uint16_t*)(ws + SZ_X + 3 * SZ_W);
  char* base = ws + SZ_X + 4 * SZ_W;
  uint16_t* Qb = (uint16_t*)(base);
  uint16_t* Kb = (uint16_t*)(base + SZ_X);
  uint16_t* Vt = (uint16_t*)(base + 2 * SZ_X);
  float*    Sb = (float*)   (base + 3 * SZ_X);
  uint16_t* Pb = (uint16_t*)(base + 3 * SZ_X + SZ_SF);
  uint16_t* Cb = (uint16_t*)(base + 3 * SZ_X + SZ_SF + SZ_SB);

  const dim3 blk(256, 1, 1);
  const float inv_sqrt_h = 0.03125f;  // 1/sqrt(1024)

  const ull sSH = (ull)SEQ * HIDDEN;
  const ull sHS = (ull)HIDDEN * SEQ;
  const ull sSS = (ull)SEQ * SEQ;

  // --- precision downcast passes (memory-bound, ~3us total at 23.3 TB/s) ---
  const int n_x8 = BATCH * SEQ * HIDDEN / 8;
  const int n_w8 = HIDDEN * HIDDEN / 8;
  cvt_f32_bf16<<<n_x8 / 256, blk, 0, stream>>>(x,  xb,  n_x8);
  cvt_f32_bf16<<<n_w8 / 256, blk, 0, stream>>>(Wq, Wqb, n_w8);
  cvt_f32_bf16<<<n_w8 / 256, blk, 0, stream>>>(Wk, Wkb, n_w8);
  cvt_f32_bf16<<<n_w8 / 256, blk, 0, stream>>>(Wv, Wvb, n_w8);
  cvt_f32_bf16<<<n_w8 / 256, blk, 0, stream>>>(Wo, Wob, n_w8);

  // --- QKV projections: per batch M=SEQ, N=HIDDEN, K=HIDDEN ---
  const dim3 gp(SEQ / 64, HIDDEN / 256, BATCH);
  gemm_bf16_nt<false, false, true><<<gp, blk, 0, stream>>>(
      xb, Wqb, bq, Qb, HIDDEN, HIDDEN, 1.0f, sSH, 0ull, sSH);
  gemm_bf16_nt<false, false, true><<<gp, blk, 0, stream>>>(
      xb, Wkb, bk, Kb, HIDDEN, HIDDEN, 1.0f, sSH, 0ull, sSH);
  // V stored transposed: Vt[b][h][s] (ldc = SEQ) so ctx GEMM is NT too.
  gemm_bf16_nt<false, true, true><<<gp, blk, 0, stream>>>(
      xb, Wvb, bv, Vt, HIDDEN, SEQ, 1.0f, sSH, 0ull, sHS);

  // --- scores = (Q @ K^T) / sqrt(H) : per batch M=SEQ, N=SEQ, K=HIDDEN ---
  const dim3 gs(SEQ / 64, SEQ / 256, BATCH);
  gemm_bf16_nt<true, false, false><<<gs, blk, 0, stream>>>(
      Qb, Kb, nullptr, Sb, HIDDEN, SEQ, inv_sqrt_h, sSH, sSH, sSS);

  // --- row softmax -> bf16 probabilities ---
  softmax_rows<<<BATCH * SEQ, blk, 0, stream>>>(Sb, Pb);

  // --- ctx = P @ V : per batch M=SEQ, N=HIDDEN, K=SEQ ---
  const dim3 gc(SEQ / 64, HIDDEN / 256, BATCH);
  gemm_bf16_nt<false, false, false><<<gc, blk, 0, stream>>>(
      Pb, Vt, nullptr, Cb, SEQ, HIDDEN, 1.0f, sSS, sHS, sSH);

  // --- output projection: out = ctx @ Wo^T + bo (f32 out) ---
  gemm_bf16_nt<true, false, true><<<gc, blk, 0, stream>>>(
      Cb, Wob, bo, out, HIDDEN, HIDDEN, 1.0f, sSH, 0ull, sSH);
}
